// ConditionedS4DPLRTransition_30305289240699
// MI455X (gfx1250) — compile-verified, matrix operand order checked by reading, not run on hardware
//
#include <hip/hip_runtime.h>
#include <hip/hip_bf16.h>
#include <stdint.h>

// ---------------- types / WMMA helpers (gfx1250, wave32) ----------------
typedef __attribute__((ext_vector_type(16))) __bf16 v16bf;
typedef __attribute__((ext_vector_type(8)))  float  v8f;

#define BATCH   4096
#define TSTEPS  32
#define DYN     256
#define STAT    128
#define UDIM    16
#define RRANK   8
#define DCPLX   128          // D = DYN/2
#define NOBS    24
#define HZDIM   384          // HZ = DYN+STAT
#define SELIN   401
#define KP0     416          // SELIN padded to mult of 32
#define KT0     13           // KP0/32
#define NHEAD   14976        // total head output columns
#define HKT     12           // HZDIM/32
#define HNT     936          // NHEAD/16
#define MROWS   32           // rows per block in heads kernel (2 M-tiles)

__device__ __forceinline__ v8f wmma_bf16(v16bf a, v16bf b, v8f c) {
    return __builtin_amdgcn_wmma_f32_16x16x32_bf16(false, a, false, b, (short)0, c, false, false);
}

// A fragment (16x32 bf16) gathered from a row-major bf16 tile with given row stride.
__device__ __forceinline__ v16bf loadA_rm(const __bf16* tile, int stride, int kt, int lane) {
    int m = lane & 15, half = lane >> 4;
    const __bf16* base = tile + m * stride + kt * 32;
    v16bf f;
#pragma unroll
    for (int p = 0; p < 8; ++p) {
        int k0 = (p < 4) ? (half * 8 + 2 * p) : (16 + half * 8 + 2 * (p - 4));
        f[2 * p]     = base[k0];
        f[2 * p + 1] = base[k0 + 1];
    }
    return f;
}

// B fragment from pre-packed weights: contiguous 32 bytes per lane.
__device__ __forceinline__ v16bf loadB_frag(const __bf16* Wp, int nt, int kt, int KT, int lane) {
    return *reinterpret_cast<const v16bf*>(Wp + (((size_t)nt * KT + kt) * 32 + lane) * 16);
}

// CDNA5 async global->LDS copy (ASYNCcnt path) + wait.
__device__ __forceinline__ void async_copy_b128(unsigned int lds_addr, const void* gptr) {
    asm volatile("global_load_async_to_lds_b128 %0, %1, off"
                 :: "v"(lds_addr), "v"(gptr) : "memory");
}
__device__ __forceinline__ void wait_async0() {
    asm volatile("s_wait_asynccnt 0x0" ::: "memory");
}

// ---------------- one-time packing kernels ----------------
__device__ __forceinline__ void unpack_idx(int idx, int KT, int& nt, int& kt, int& k, int& n) {
    int e = idx & 15, lane = (idx >> 4) & 31, rem = idx >> 9;
    kt = rem % KT; nt = rem / KT;
    int half = lane >> 4, p = e >> 1, j = e & 1;
    k = kt * 32 + ((p < 4) ? (half * 8 + 2 * p + j) : (16 + half * 8 + 2 * (p - 4) + j));
    n = nt * 16 + (lane & 15);
}

__global__ void pack_w(const float* __restrict__ W, __bf16* __restrict__ out,
                       int K, int N, int KT, int NT) {
    int idx = blockIdx.x * blockDim.x + threadIdx.x;
    int total = NT * KT * 512;
    if (idx >= total) return;
    int nt, kt, gk, gn;
    unpack_idx(idx, KT, nt, kt, gk, gn);
    float v = (gk < K && gn < N) ? W[(size_t)gk * N + gn] : 0.0f;
    out[idx] = (__bf16)v;
}

__global__ void pack_heads(const float* Wa, const float* Wom, const float* Wur, const float* Wui,
                           const float* Wvr, const float* Wvi, const float* Wbr, const float* Wbi,
                           const float* Wcr, const float* Wci, const float* Wdt,
                           __bf16* __restrict__ out) {
    int idx = blockIdx.x * blockDim.x + threadIdx.x;
    int total = HNT * HKT * 512;
    if (idx >= total) return;
    int nt, kt, gk, gn;
    unpack_idx(idx, HKT, nt, kt, gk, gn);
    const int offs[12] = {0,128,256,1280,2304,3328,4352,6400,8448,11520,14592,14976};
    const float* Ws[11] = {Wa,Wom,Wur,Wui,Wvr,Wvi,Wbr,Wbi,Wcr,Wci,Wdt};
    int h = 0;
    while (gn >= offs[h + 1]) ++h;
    int wdt = offs[h + 1] - offs[h];
    out[idx] = (__bf16)Ws[h][(size_t)gk * wdt + (gn - offs[h])];
}

__global__ void pack_bias(const float* ba, const float* bo, const float* bur, const float* bui,
                          const float* bvr, const float* bvi, const float* bbr, const float* bbi,
                          const float* bcr, const float* bci, const float* bdt,
                          float* __restrict__ out) {
    int c = blockIdx.x * blockDim.x + threadIdx.x;
    if (c >= NHEAD) return;
    const int offs[12] = {0,128,256,1280,2304,3328,4352,6400,8448,11520,14592,14976};
    const float* Bs[11] = {ba,bo,bur,bui,bvr,bvi,bbr,bbi,bcr,bci,bdt};
    int h = 0;
    while (c >= offs[h + 1]) ++h;
    out[c] = Bs[h][c - offs[h]];
}

__global__ void init_z(const float* __restrict__ z_dyn,
                       float* __restrict__ zr, float* __restrict__ zi) {
    int idx = blockIdx.x * blockDim.x + threadIdx.x;
    if (idx >= BATCH * DCPLX) return;
    int b = idx >> 7, d = idx & 127;
    zr[idx] = z_dyn[(size_t)b * DYN + d];
    zi[idx] = z_dyn[(size_t)b * DYN + DCPLX + d];
}

__global__ void zero_stats(float* __restrict__ stats) {
    if (threadIdx.x < 512) stats[threadIdx.x] = 0.0f;
}

// ---------------- per-step selector kernels ----------------
__global__ __launch_bounds__(256) void sel0(const float* __restrict__ zr, const float* __restrict__ zi,
                                            const float* __restrict__ zs, const float* __restrict__ Uall,
                                            const float* __restrict__ dtv, int t,
                                            const __bf16* __restrict__ W0P, const float* __restrict__ sb0,
                                            float* __restrict__ pre0, float* __restrict__ stats) {
    __shared__ __bf16 xT[16][KP0];
    int tid = threadIdx.x, row0 = blockIdx.x * 16;
    for (int idx = tid; idx < 16 * KP0; idx += 256) {
        int r = idx / KP0, c = idx % KP0;
        int b = row0 + r;
        float v;
        if      (c < 128) v = zr[(size_t)b * 128 + c];
        else if (c < 256) v = zi[(size_t)b * 128 + (c - 128)];
        else if (c < 384) v = zs[(size_t)b * 128 + (c - 256)];
        else if (c < 400) v = Uall[((size_t)t * BATCH + b) * UDIM + (c - 384)];
        else if (c == 400) v = dtv[b];
        else v = 0.0f;
        xT[r][c] = (__bf16)v;
    }
    __syncthreads();
    int lane = tid & 31, w = tid >> 5;
    int nt = w;
    v8f acc = {};
#pragma unroll
    for (int kt = 0; kt < KT0; ++kt) {
        v16bf a = loadA_rm(&xT[0][0], KP0, kt, lane);
        v16bf b = loadB_frag(W0P, nt, kt, KT0, lane);
        acc = wmma_bf16(a, b, acc);
    }
    int n = nt * 16 + (lane & 15), half = lane >> 4;
    float bb = sb0[n], s = 0.f, s2 = 0.f;
#pragma unroll
    for (int i = 0; i < 8; ++i) {
        float v = acc[i] + bb;
        pre0[(size_t)(row0 + i + 8 * half) * 128 + n] = v;
        s += v; s2 += v * v;
    }
    atomicAdd(&stats[n], s);
    atomicAdd(&stats[128 + n], s2);
}

__global__ __launch_bounds__(256) void sel1(const float* __restrict__ pre0, const float* __restrict__ statsIn,
                                            const float* __restrict__ g, const float* __restrict__ bB,
                                            const __bf16* __restrict__ W1P, const float* __restrict__ sb1,
                                            float* __restrict__ pre1, float* __restrict__ statsOut) {
    __shared__ __bf16 hT[16][128];
    __shared__ float scale[128], shift[128];
    int tid = threadIdx.x, row0 = blockIdx.x * 16;
    if (tid < 128) {
        float m = statsIn[tid] * (1.0f / BATCH);
        float v = statsIn[128 + tid] * (1.0f / BATCH) - m * m;
        float sc = rsqrtf(v + 1e-5f) * g[tid];
        scale[tid] = sc;
        shift[tid] = bB[tid] - m * sc;
    }
    __syncthreads();
    for (int idx = tid; idx < 16 * 128; idx += 256) {
        int r = idx >> 7, c = idx & 127;
        float v = pre0[(size_t)(row0 + r) * 128 + c] * scale[c] + shift[c];
        hT[r][c] = (__bf16)fmaxf(v, 0.0f);
    }
    __syncthreads();
    int lane = tid & 31, w = tid >> 5, nt = w;
    v8f acc = {};
#pragma unroll
    for (int kt = 0; kt < 4; ++kt) {
        v16bf a = loadA_rm(&hT[0][0], 128, kt, lane);
        v16bf b = loadB_frag(W1P, nt, kt, 4, lane);
        acc = wmma_bf16(a, b, acc);
    }
    int n = nt * 16 + (lane & 15), half = lane >> 4;
    float bb = sb1[n], s = 0.f, s2 = 0.f;
#pragma unroll
    for (int i = 0; i < 8; ++i) {
        float v = acc[i] + bb;
        pre1[(size_t)(row0 + i + 8 * half) * 128 + n] = v;
        s += v; s2 += v * v;
    }
    atomicAdd(&statsOut[n], s);
    atomicAdd(&statsOut[128 + n], s2);
}

__global__ __launch_bounds__(256) void sel2(const float* __restrict__ pre1, const float* __restrict__ statsIn,
                                            const float* __restrict__ g, const float* __restrict__ bB,
                                            const __bf16* __restrict__ WoP, const float* __restrict__ sbo,
                                            __bf16* __restrict__ hz) {
    __shared__ __bf16 hT[16][128];
    __shared__ float scale[128], shift[128];
    int tid = threadIdx.x, row0 = blockIdx.x * 16;
    if (tid < 128) {
        float m = statsIn[tid] * (1.0f / BATCH);
        float v = statsIn[128 + tid] * (1.0f / BATCH) - m * m;
        float sc = rsqrtf(v + 1e-5f) * g[tid];
        scale[tid] = sc;
        shift[tid] = bB[tid] - m * sc;
    }
    __syncthreads();
    for (int idx = tid; idx < 16 * 128; idx += 256) {
        int r = idx >> 7, c = idx & 127;
        float v = pre1[(size_t)(row0 + r) * 128 + c] * scale[c] + shift[c];
        hT[r][c] = (__bf16)fmaxf(v, 0.0f);
    }
    __syncthreads();
    int lane = tid & 31, w = tid >> 5, half = lane >> 4;
#pragma unroll
    for (int rep = 0; rep < 3; ++rep) {
        int nt = w + rep * 8;
        v8f acc = {};
#pragma unroll
        for (int kt = 0; kt < 4; ++kt) {
            v16bf a = loadA_rm(&hT[0][0], 128, kt, lane);
            v16bf b = loadB_frag(WoP, nt, kt, 4, lane);
            acc = wmma_bf16(a, b, acc);
        }
        int n = nt * 16 + (lane & 15);
        float bb = sbo[n];
#pragma unroll
        for (int i = 0; i < 8; ++i)
            hz[(size_t)(row0 + i + 8 * half) * HZDIM + n] = (__bf16)(acc[i] + bb);
    }
}

// ---------------- fused heads GEMM + DPLR recurrence (2 M-tiles per wave) ----------------
__device__ __forceinline__ void head_tile2(const v16bf* A0, const v16bf* A1,
                                           const __bf16* __restrict__ WhP, int nt, int lane,
                                           v8f& acc0, v8f& acc1) {
    if (nt + 8 < HNT)
        __builtin_prefetch(WhP + ((size_t)(nt + 8) * HKT * 32) * 16, 0, 1);
    acc0 = (v8f){}; acc1 = (v8f){};
#pragma unroll
    for (int kt = 0; kt < HKT; ++kt) {
        v16bf b = loadB_frag(WhP, nt, kt, HKT, lane);   // reused for both M-tiles
        acc0 = wmma_bf16(A0[kt], b, acc0);
        acc1 = wmma_bf16(A1[kt], b, acc1);
    }
}

__global__ __launch_bounds__(256) void heads_step(const __bf16* __restrict__ hz,
                                                  const __bf16* __restrict__ WhP,
                                                  const float* __restrict__ bh,
                                                  const float* __restrict__ Uall,
                                                  const float* __restrict__ dtv, int t,
                                                  float* __restrict__ zr, float* __restrict__ zi,
                                                  float* __restrict__ Zout, float* __restrict__ Yout) {
    // ZR/ZI hold z during phases 0-2, overwritten in place with zn afterwards.
    __shared__ float ZR[MROWS][128], ZI[MROWS][128];
    __shared__ float sR[MROWS][RRANK], sI[MROWS][RRANK];
    __shared__ float yT[MROWS][NOBS];
    __shared__ float udtT[MROWS][UDIM];

    int tid = threadIdx.x, lane = tid & 31, w = tid >> 5, row0 = blockIdx.x * MROWS;
    int nl = lane & 15, half = lane >> 4;

    // ---- phase 0: async-stage z carry into LDS; stage u*dt; zero accumulators ----
    {
        unsigned int ldsR = (unsigned int)(uintptr_t)(&ZR[0][0]);
        unsigned int ldsI = (unsigned int)(uintptr_t)(&ZI[0][0]);
#pragma unroll
        for (int it = 0; it < 4; ++it) {
            int idx = tid + it * 256;                    // dword4 index
            async_copy_b128(ldsR + idx * 16, zr + (size_t)row0 * 128 + idx * 4);
            async_copy_b128(ldsI + idx * 16, zi + (size_t)row0 * 128 + idx * 4);
        }
    }
    for (int idx = tid; idx < MROWS * UDIM; idx += 256) {
        int r = idx >> 4, u = idx & 15;
        udtT[r][u] = Uall[((size_t)t * BATCH + row0 + r) * UDIM + u] * dtv[row0 + r];
    }
    if (tid < MROWS * RRANK) { sR[tid >> 3][tid & 7] = 0.f; sI[tid >> 3][tid & 7] = 0.f; }
    for (int idx = tid; idx < MROWS * NOBS; idx += 256) yT[idx / NOBS][idx % NOBS] = 0.f;

    // A fragments for both 16-row tiles (kept resident: 2x reuse of every B fragment)
    v16bf A0[HKT], A1[HKT];
#pragma unroll
    for (int mt = 0; mt < 2; ++mt) {
        const __bf16* base = hz + (size_t)(row0 + mt * 16 + nl) * HZDIM;
        v16bf* A = mt ? A1 : A0;
#pragma unroll
        for (int kt = 0; kt < HKT; ++kt) {
#pragma unroll
            for (int p = 0; p < 8; ++p) {
                int k0 = kt * 32 + ((p < 4) ? (half * 8 + 2 * p) : (16 + half * 8 + 2 * (p - 4)));
                A[kt][2 * p]     = base[k0];
                A[kt][2 * p + 1] = base[k0 + 1];
            }
        }
    }
    wait_async0();
    __syncthreads();

    // ---- phase 1: alpha (nt=w) + omega (nt=w+8) kept in regs; V tiles -> s = V^H z ----
    v8f aA0, aA1, aO0, aO1;
    head_tile2(A0, A1, WhP, w,     lane, aA0, aA1);
    head_tile2(A0, A1, WhP, w + 8, lane, aO0, aO1);

    for (int nt = 144 + w; nt < 272; nt += 8) {
        v8f acc0, acc1;
        head_tile2(A0, A1, WhP, nt, lane, acc0, acc1);
        int c = nt * 16 + nl; float bb = bh[c];
        int local = c - 2304;
        bool isVr = local < 1024; int l2 = isVr ? local : local - 1024;
        int d = l2 >> 3, r = l2 & 7;
#pragma unroll
        for (int mt = 0; mt < 2; ++mt) {
            v8f acc = mt ? acc1 : acc0;
#pragma unroll
            for (int i = 0; i < 8; ++i) {
                float v = acc[i] + bb; int m = mt * 16 + i + 8 * half;
                if (isVr) { atomicAdd(&sR[m][r],  v * ZR[m][d]); atomicAdd(&sI[m][r],  v * ZI[m][d]); }
                else      { atomicAdd(&sR[m][r],  v * ZI[m][d]); atomicAdd(&sI[m][r], -v * ZR[m][d]); }
            }
        }
    }
    __syncthreads();

    // ---- phase 2: Lambda = exp(-softplus(a)+i w); zn = Lambda*z written in place ----
    {
        int c = w * 16 + nl;                       // d index (alpha col), 0..127
        float bbA = bh[c], bbO = bh[128 + c];
#pragma unroll
        for (int mt = 0; mt < 2; ++mt) {
            v8f aA = mt ? aA1 : aA0;
            v8f aO = mt ? aO1 : aO0;
#pragma unroll
            for (int i = 0; i < 8; ++i) {
                int m = mt * 16 + i + 8 * half;
                float a  = aA[i] + bbA;
                float om = aO[i] + bbO;
                float lam = 1.0f / (1.0f + __expf(a));      // exp(-softplus(a))
                float sn, cs; __sincosf(om, &sn, &cs);
                float LR = lam * cs, LI = lam * sn;
                float zrv = ZR[m][c], ziv = ZI[m][c];
                ZR[m][c] = LR * zrv - LI * ziv;
                ZI[m][c] = LR * ziv + LI * zrv;
            }
        }
    }
    __syncthreads();

    // ---- phase 3: U tiles (zn += U s) and B tiles (zn += B*(u dt)) ----
    for (int nt = 16 + w; nt < 144; nt += 8) {
        v8f acc0, acc1;
        head_tile2(A0, A1, WhP, nt, lane, acc0, acc1);
        int c = nt * 16 + nl; float bb = bh[c];
        int local = c - 256;
        bool isUr = local < 1024; int l2 = isUr ? local : local - 1024;
        int d = l2 >> 3, r = l2 & 7;
#pragma unroll
        for (int mt = 0; mt < 2; ++mt) {
            v8f acc = mt ? acc1 : acc0;
#pragma unroll
            for (int i = 0; i < 8; ++i) {
                float v = acc[i] + bb; int m = mt * 16 + i + 8 * half;
                float srv = sR[m][r], siv = sI[m][r];
                if (isUr) { atomicAdd(&ZR[m][d],  v * srv); atomicAdd(&ZI[m][d], v * siv); }
                else      { atomicAdd(&ZR[m][d], -v * siv); atomicAdd(&ZI[m][d], v * srv); }
            }
        }
    }
    for (int nt = 272 + w; nt < 528; nt += 8) {
        v8f acc0, acc1;
        head_tile2(A0, A1, WhP, nt, lane, acc0, acc1);
        int c = nt * 16 + nl; float bb = bh[c];
        int local = c - 4352;
        bool isBr = local < 2048; int l2 = isBr ? local : local - 2048;
        int d = l2 >> 4, u = l2 & 15;
#pragma unroll
        for (int mt = 0; mt < 2; ++mt) {
            v8f acc = mt ? acc1 : acc0;
#pragma unroll
            for (int i = 0; i < 8; ++i) {
                float v = acc[i] + bb; int m = mt * 16 + i + 8 * half;
                float uv = udtT[m][u];
                if (isBr) atomicAdd(&ZR[m][d], v * uv); else atomicAdd(&ZI[m][d], v * uv);
            }
        }
    }
    __syncthreads();

    // ---- phase 4: write Z & carry; C tiles (y += Re(C zn)); D tiles (y += D*(u dt)) ----
    for (int idx = tid; idx < MROWS * 256; idx += 256) {
        int r = idx >> 8, c = idx & 255;
        int b = row0 + r;
        float v = (c < 128) ? ZR[r][c] : ZI[r][c - 128];
        Zout[((size_t)t * BATCH + b) * DYN + c] = v;
        if (c < 128) zr[(size_t)b * 128 + c] = v;
        else         zi[(size_t)b * 128 + (c - 128)] = v;
    }
    for (int nt = 528 + w; nt < 912; nt += 8) {
        v8f acc0, acc1;
        head_tile2(A0, A1, WhP, nt, lane, acc0, acc1);
        int c = nt * 16 + nl; float bb = bh[c];
        int local = c - 8448;
        bool isCr = local < 3072; int l2 = isCr ? local : local - 3072;
        int n = l2 >> 7, d = l2 & 127;
#pragma unroll
        for (int mt = 0; mt < 2; ++mt) {
            v8f acc = mt ? acc1 : acc0;
#pragma unroll
            for (int i = 0; i < 8; ++i) {
                float v = acc[i] + bb; int m = mt * 16 + i + 8 * half;
                if (isCr) atomicAdd(&yT[m][n],  v * ZR[m][d]);
                else      atomicAdd(&yT[m][n], -v * ZI[m][d]);
            }
        }
    }
    for (int nt = 912 + w; nt < 936; nt += 8) {
        v8f acc0, acc1;
        head_tile2(A0, A1, WhP, nt, lane, acc0, acc1);
        int c = nt * 16 + nl; float bb = bh[c];
        int local = c - 14592;
        int n = local >> 4, u = local & 15;
#pragma unroll
        for (int mt = 0; mt < 2; ++mt) {
            v8f acc = mt ? acc1 : acc0;
#pragma unroll
            for (int i = 0; i < 8; ++i) {
                float v = acc[i] + bb; int m = mt * 16 + i + 8 * half;
                atomicAdd(&yT[m][n], v * udtT[m][u]);
            }
        }
    }
    __syncthreads();
    for (int idx = tid; idx < MROWS * NOBS; idx += 256) {
        int r = idx / NOBS, n = idx % NOBS;
        Yout[((size_t)t * BATCH + row0 + r) * NOBS + n] = yT[r][n];
    }
}

// ---------------- host side ----------------
extern "C" void kernel_launch(void* const* d_in, const int* in_sizes, int n_in,
                              void* d_out, int out_size, void* d_ws, size_t ws_size,
                              hipStream_t stream) {
    (void)in_sizes; (void)n_in; (void)out_size; (void)ws_size;
    const float* z_dyn    = (const float*)d_in[0];
    const float* z_static = (const float*)d_in[1];
    const float* dtv      = (const float*)d_in[2];
    const float* Uall     = (const float*)d_in[3];
    const float* sW0 = (const float*)d_in[4];  const float* sb0 = (const float*)d_in[5];
    const float* sg0 = (const float*)d_in[6];  const float* sB0 = (const float*)d_in[7];
    const float* sW1 = (const float*)d_in[8];  const float* sb1 = (const float*)d_in[9];
    const float* sg1 = (const float*)d_in[10]; const float* sB1 = (const float*)d_in[11];
    const float* sWo = (const float*)d_in[12]; const float* sbo = (const float*)d_in[13];
    const float* W_alpha=(const float*)d_in[14], *b_alpha=(const float*)d_in[15];
    const float* W_omega=(const float*)d_in[16], *b_omega=(const float*)d_in[17];
    const float* W_Ur=(const float*)d_in[18],    *b_Ur=(const float*)d_in[19];
    const float* W_Ui=(const float*)d_in[20],    *b_Ui=(const float*)d_in[21];
    const float* W_Vr=(const float*)d_in[22],    *b_Vr=(const float*)d_in[23];
    const float* W_Vi=(const float*)d_in[24],    *b_Vi=(const float*)d_in[25];
    const float* W_Br=(const float*)d_in[26],    *b_Br=(const float*)d_in[27];
    const float* W_Bi=(const float*)d_in[28],    *b_Bi=(const float*)d_in[29];
    const float* W_Cr=(const float*)d_in[30],    *b_Cr=(const float*)d_in[31];
    const float* W_Ci=(const float*)d_in[32],    *b_Ci=(const float*)d_in[33];
    const float* W_Dt=(const float*)d_in[34],    *b_Dt=(const float*)d_in[35];

    float* Zout = (float*)d_out;
    float* Yout = Zout + (size_t)TSTEPS * BATCH * DYN;

    char* base = (char*)d_ws;
    size_t off = 0;
    auto carve = [&](size_t bytes) -> char* {
        off = (off + 255) & ~(size_t)255;
        char* p = base + off;
        off += bytes;
        return p;
    };
    __bf16* WhP  = (__bf16*)carve((size_t)HNT * HKT * 512 * 2);
    __bf16* W0P  = (__bf16*)carve((size_t)8  * KT0 * 512 * 2);
    __bf16* W1P  = (__bf16*)carve((size_t)8  * 4   * 512 * 2);
    __bf16* WoP  = (__bf16*)carve((size_t)24 * 4   * 512 * 2);
    float*  bh   = (float*) carve((size_t)NHEAD * 4);
    float*  zrC  = (float*) carve((size_t)BATCH * 128 * 4);
    float*  ziC  = (float*) carve((size_t)BATCH * 128 * 4);
    float*  pre0 = (float*) carve((size_t)BATCH * 128 * 4);
    float*  pre1 = (float*) carve((size_t)BATCH * 128 * 4);
    __bf16* hz   = (__bf16*)carve((size_t)BATCH * HZDIM * 2);
    float*  stats= (float*) carve(512 * 4);

    {
        int totH = HNT * HKT * 512;
        pack_heads<<<(totH + 255) / 256, 256, 0, stream>>>(
            W_alpha, W_omega, W_Ur, W_Ui, W_Vr, W_Vi, W_Br, W_Bi, W_Cr, W_Ci, W_Dt, WhP);
        pack_bias<<<(NHEAD + 255) / 256, 256, 0, stream>>>(
            b_alpha, b_omega, b_Ur, b_Ui, b_Vr, b_Vi, b_Br, b_Bi, b_Cr, b_Ci, b_Dt, bh);
        int t0 = 8 * KT0 * 512;
        pack_w<<<(t0 + 255) / 256, 256, 0, stream>>>(sW0, W0P, SELIN, 128, KT0, 8);
        int t1 = 8 * 4 * 512;
        pack_w<<<(t1 + 255) / 256, 256, 0, stream>>>(sW1, W1P, 128, 128, 4, 8);
        int t2 = 24 * 4 * 512;
        pack_w<<<(t2 + 255) / 256, 256, 0, stream>>>(sWo, WoP, 128, HZDIM, 4, 24);
        init_z<<<(BATCH * 128 + 255) / 256, 256, 0, stream>>>(z_dyn, zrC, ziC);
    }

    dim3 gridSel(BATCH / 16), gridHeads(BATCH / MROWS), block(256);
    for (int t = 0; t < TSTEPS; ++t) {
        zero_stats<<<1, 512, 0, stream>>>(stats);
        sel0<<<gridSel, block, 0, stream>>>(zrC, ziC, z_static, Uall, dtv, t, W0P, sb0, pre0, stats);
        sel1<<<gridSel, block, 0, stream>>>(pre0, stats, sg0, sB0, W1P, sb1, pre1, stats + 256);
        sel2<<<gridSel, block, 0, stream>>>(pre1, stats + 256, sg1, sB1, WoP, sbo, hz);
        heads_step<<<gridHeads, block, 0, stream>>>(hz, WhP, bh, Uall, dtv, t, zrC, ziC, Zout, Yout);
    }
}
